// Kistmat_AI_51445118271951
// MI455X (gfx1250) — matrix operand order repaired and checked by reading, not compiled, not run on hardware
//
// MI455X (gfx1250) implementation of the LSTM + memory-read network.
// f16 WMMA (v_wmma_f32_16x16x32_f16) for all GEMMs, f32 accumulation.
// Attention-over-length-1 simplified away (softmax of singleton == 1 -> ctx == v).
// GEMM: 128x128x32 tiles, LDS double-buffered, bounds checks compiled out for
// the aligned (M%128==0 && N%128==0) case that covers all heavy GEMMs.
#include <hip/hip_runtime.h>
#include <hip/hip_bf16.h>
#include <stdint.h>

typedef __attribute__((ext_vector_type(16))) _Float16 v16h;
typedef __attribute__((ext_vector_type(8)))  float    v8f;

#define LDSS 40   // LDS row stride in f16 elems (32 + 8 pad), keeps 16B alignment

// ---------------------------------------------------------------- GEMM (WMMA)
// C[M,N] = A[M,K](f16) * Bt[N,K](f16)^T + bias[n] + add[m,n](f16)
// Block: 256 thr = 8 waves; tile 128x128x32; waves 4(M)x2(N); per-wave 32x64.
struct GemmP {
  const _Float16* A;   long long ldA, sA;
  const _Float16* Bt;  long long ldB, sB;
  const float*    bias; long long sBias;
  const _Float16* add; long long ldAdd, sAdd;
  float*          Cf;  long long sCf;
  _Float16*       Ch;  long long sCh;
  long long ldC;
  int M, N, K;
};

template <bool ALIGNED>
__global__ __launch_bounds__(256) void gemm_f16_wmma(GemmP p) {
  __shared__ __attribute__((aligned(16))) _Float16 As[2][128 * LDSS];
  __shared__ __attribute__((aligned(16))) _Float16 Bs[2][128 * LDSS];

  const int z = blockIdx.z;
  const _Float16* A   = p.A  + (long long)z * p.sA;
  const _Float16* Bt  = p.Bt + (long long)z * p.sB;
  const float*    bias = p.bias ? p.bias + (long long)z * p.sBias : nullptr;
  const _Float16* add  = p.add  ? p.add  + (long long)z * p.sAdd  : nullptr;
  float*    Cf = p.Cf ? p.Cf + (long long)z * p.sCf : nullptr;
  _Float16* Ch = p.Ch ? p.Ch + (long long)z * p.sCh : nullptr;

  const int m0 = blockIdx.y * 128;
  const int n0 = blockIdx.x * 128;
  const int tid  = threadIdx.x;
  const int wave = tid >> 5, lane = tid & 31;
  const int wm = wave & 3, wn = wave >> 2;       // 4 waves along M, 2 along N
  const int lm = lane & 15, kh = lane >> 4;      // lane-row, lane-half

  // ---- global loader: thread covers rows r and r+64 at 8-elem chunk `seg`
  const int lrow = tid >> 2;                     // 0..63
  const int seg  = tid & 3;                      // 0..3
  const _Float16* ga0 = A  + (long long)(m0 + lrow)      * p.ldA + seg * 8;
  const _Float16* ga1 = A  + (long long)(m0 + lrow + 64) * p.ldA + seg * 8;
  const _Float16* gb0 = Bt + (long long)(n0 + lrow)      * p.ldB + seg * 8;
  const _Float16* gb1 = Bt + (long long)(n0 + lrow + 64) * p.ldB + seg * 8;
  const bool okA0 = ALIGNED || (m0 + lrow      < p.M);
  const bool okA1 = ALIGNED || (m0 + lrow + 64 < p.M);
  const bool okB0 = ALIGNED || (n0 + lrow      < p.N);
  const bool okB1 = ALIGNED || (n0 + lrow + 64 < p.N);
  const int lds_off0 = lrow * LDSS + seg * 8;
  const int lds_off1 = (lrow + 64) * LDSS + seg * 8;

  uint4 ra0, ra1, rb0, rb1;
  const uint4 zv = make_uint4(0u, 0u, 0u, 0u);

  // ---- LDS read bases for the WMMA fragments
  const _Float16* arow0 = nullptr;  // set per-buffer below via offsets
  const int aoff0 = (wm * 32 + lm) * LDSS;            // m-subtile 0
  const int aoff1 = (wm * 32 + 16 + lm) * LDSS;       // m-subtile 1
  const int boffj = (wn * 64 + lm) * LDSS;            // n-subtile j adds j*16*LDSS
  (void)arow0;

  v8f acc[2][4];
#pragma unroll
  for (int i = 0; i < 2; ++i)
#pragma unroll
    for (int j = 0; j < 4; ++j)
#pragma unroll
      for (int r = 0; r < 8; ++r) acc[i][j][r] = 0.0f;

  const int nk = p.K >> 5;   // K is always a multiple of 32 here

  // prologue: fetch tile 0 and store to buffer 0
  ra0 = okA0 ? *(const uint4*)ga0 : zv;
  ra1 = okA1 ? *(const uint4*)ga1 : zv;
  rb0 = okB0 ? *(const uint4*)gb0 : zv;
  rb1 = okB1 ? *(const uint4*)gb1 : zv;
  *(uint4*)(&As[0][lds_off0]) = ra0;
  *(uint4*)(&As[0][lds_off1]) = ra1;
  *(uint4*)(&Bs[0][lds_off0]) = rb0;
  *(uint4*)(&Bs[0][lds_off1]) = rb1;
  ga0 += 32; ga1 += 32; gb0 += 32; gb1 += 32;

  for (int kt = 0; kt < nk; ++kt) {
    __syncthreads();   // buffer (kt&1) ready; other buffer free to overwrite
    const int cur = kt & 1, nxt = cur ^ 1;
    const bool more = (kt + 1 < nk);

    // prefetch next tile into registers (overlaps with WMMA below)
    if (more) {
      ra0 = okA0 ? *(const uint4*)ga0 : zv;
      ra1 = okA1 ? *(const uint4*)ga1 : zv;
      rb0 = okB0 ? *(const uint4*)gb0 : zv;
      rb1 = okB1 ? *(const uint4*)gb1 : zv;
      ga0 += 32; ga1 += 32; gb0 += 32; gb1 += 32;
    }

    // fragments per CDNA5 ISA 16-bit layouts (2x ds_load_b128 each)
    v16h afr[2], bfr[4];
    {
      const _Float16* b0 = &As[cur][aoff0];
      const _Float16* b1 = &As[cur][aoff1];
      ((uint4*)&afr[0])[0] = *(const uint4*)(b0 + 8 * kh);       // K = 8h..8h+7
      ((uint4*)&afr[0])[1] = *(const uint4*)(b0 + 16 + 8 * kh);  // K = 16+8h..
      ((uint4*)&afr[1])[0] = *(const uint4*)(b1 + 8 * kh);
      ((uint4*)&afr[1])[1] = *(const uint4*)(b1 + 16 + 8 * kh);
    }
#pragma unroll
    for (int j = 0; j < 4; ++j) {
      const _Float16* base = &Bs[cur][boffj + j * 16 * LDSS];
      ((uint4*)&bfr[j])[0] = *(const uint4*)(base + 8 * kh);     // K = 8h..8h+7
      ((uint4*)&bfr[j])[1] = *(const uint4*)(base + 16 + 8 * kh);
    }
#pragma unroll
    for (int i = 0; i < 2; ++i)
#pragma unroll
      for (int j = 0; j < 4; ++j)
        acc[i][j] = __builtin_amdgcn_wmma_f32_16x16x32_f16(
            false, afr[i], false, bfr[j], (short)0, acc[i][j], false, false);

    // stage next tile into the alternate LDS buffer
    if (more) {
      *(uint4*)(&As[nxt][lds_off0]) = ra0;
      *(uint4*)(&As[nxt][lds_off1]) = ra1;
      *(uint4*)(&Bs[nxt][lds_off0]) = rb0;
      *(uint4*)(&Bs[nxt][lds_off1]) = rb1;
    }
  }

  // Epilogue: D layout — lane n = lm, vgpr r -> m = 8*half + r
#pragma unroll
  for (int i = 0; i < 2; ++i) {
#pragma unroll
    for (int j = 0; j < 4; ++j) {
      int n = n0 + wn * 64 + j * 16 + lm;
      if (!ALIGNED && n >= p.N) continue;
      float bv = bias ? bias[n] : 0.0f;
#pragma unroll
      for (int r = 0; r < 8; ++r) {
        int m = m0 + wm * 32 + i * 16 + kh * 8 + r;
        if (!ALIGNED && m >= p.M) continue;
        float v = acc[i][j][r] + bv;
        if (add) v += (float)add[(long long)m * p.ldAdd + n];
        if (Cf) Cf[(long long)m * p.ldC + n] = v;
        if (Ch) Ch[(long long)m * p.ldC + n] = (_Float16)v;
      }
    }
  }
}

// ------------------------------------------------------------ helper kernels
__global__ void transpose_conv(const float* in, _Float16* out, int R, int C) {
  __shared__ float t[32][33];
  int c0 = blockIdx.x * 32, r0 = blockIdx.y * 32;
  int tx = threadIdx.x, ty = threadIdx.y;
  for (int i = 0; i < 32; i += 8) {
    int r = r0 + ty + i, c = c0 + tx;
    t[ty + i][tx] = (r < R && c < C) ? in[(long long)r * C + c] : 0.0f;
  }
  __syncthreads();
  for (int i = 0; i < 32; i += 8) {
    int c = c0 + ty + i, r = r0 + tx;
    if (c < C && r < R) out[(long long)c * R + r] = (_Float16)t[tx][ty + i];
  }
}

__global__ void embed_kernel(const int* tok, const float* emb, _Float16* xh, int rows) {
  int idx = blockIdx.x * 256 + threadIdx.x;
  if (idx >= rows * 64) return;
  int row = idx >> 6, d = idx & 63;
  xh[idx] = (_Float16)emb[(long long)tok[row] * 64 + d];
}

__global__ void zero_u32(uint32_t* p, int n) {
  int i = blockIdx.x * 256 + threadIdx.x;
  if (i < n) p[i] = 0u;
}

__device__ __forceinline__ float sigm(float x) { return 1.0f / (1.0f + __expf(-x)); }

// Z[2][512][2048] -> gates; C[2][512][512] f32 state; H[2][512][512] f16 state.
// mode 1: hs[(b*64+t)*1024 + dir*512 + j] (layer-1 sequence out, f16)
// mode 2: hs[b*1024 + dir*512 + j]        (layer-2 final state, overwritten each t)
__global__ void lstm_gate(const float* Z, float* C, _Float16* H,
                          _Float16* hs, int t0, int t1, int mode) {
  int dir = blockIdx.y;
  int idx = blockIdx.x * blockDim.x + threadIdx.x;   // 0..512*512-1
  int b = idx >> 9, j = idx & 511;
  const float* zr = Z + (long long)dir * 512 * 2048 + (long long)b * 2048;
  float ig = sigm(zr[j]);
  float fg = sigm(zr[512 + j]);
  float gg = tanhf(zr[1024 + j]);
  float og = sigm(zr[1536 + j]);
  long long cidx = (long long)dir * 512 * 512 + (long long)b * 512 + j;
  float c = fg * C[cidx] + ig * gg;
  C[cidx] = c;
  float h = og * tanhf(c);
  H[cidx] = (_Float16)h;
  int t = dir ? t1 : t0;
  if (mode == 1)      hs[((long long)(b * 64 + t)) * 1024 + dir * 512 + j] = (_Float16)h;
  else if (mode == 2) hs[(long long)b * 1024 + dir * 512 + j] = (_Float16)h;
}

// ------------------------------------------------------------ memory reads
struct MemsP { const float* K[6]; const float* V[6]; int M[6]; };

__global__ __launch_bounds__(64) void mem_read_kernel(const float* query, MemsP mp, float* reads) {
  __shared__ float q[64];
  __shared__ float red[64];
  __shared__ float sc[10016];          // max M = 10000 (ltm)
  int mem = blockIdx.y, b = blockIdx.x, tid = threadIdx.x;
  int M = mp.M[mem];
  const float* Kp = mp.K[mem];
  const float* Vp = mp.V[mem];
  q[tid] = query[(long long)b * 64 + tid];
  __syncthreads();

  float lmax = -1e30f;
  for (int m = tid; m < M; m += 64) {
    const float* kr = Kp + (long long)m * 64;
    float s = 0.0f;
    for (int kk = 0; kk < 64; kk += 4) {
      float4 kv = *(const float4*)(kr + kk);
      s += q[kk] * kv.x + q[kk + 1] * kv.y + q[kk + 2] * kv.z + q[kk + 3] * kv.w;
    }
    s *= 0.125f;                       // / sqrt(64)
    sc[m] = s;
    lmax = fmaxf(lmax, s);
  }
  red[tid] = lmax; __syncthreads();
  for (int off = 32; off > 0; off >>= 1) {
    if (tid < off) red[tid] = fmaxf(red[tid], red[tid + off]);
    __syncthreads();
  }
  float mx = red[0]; __syncthreads();

  float lsum = 0.0f;
  for (int m = tid; m < M; m += 64) { float e = __expf(sc[m] - mx); sc[m] = e; lsum += e; }
  red[tid] = lsum; __syncthreads();
  for (int off = 32; off > 0; off >>= 1) {
    if (tid < off) red[tid] += red[tid + off];
    __syncthreads();
  }
  float inv = 1.0f / red[0];

  float acc = 0.0f;
  for (int m = 0; m < M; ++m) acc += sc[m] * Vp[(long long)m * 64 + tid];
  acc *= inv;
  if (mem == 1) {                      // form read * sigmoid(sum(query[:,0:6]))
    float terms = q[0] + q[1] + q[2] + q[3] + q[4] + q[5];
    acc *= sigm(terms);
  }
  reads[((long long)mem * 512 + b) * 64 + tid] = acc;
}

// mean(reads) -> relu(@rs_W+rs_b) -> BN -> @st_W+st_b -> sigmoid(@fo_W+fo_b)
__global__ __launch_bounds__(512) void tail_kernel(
    const float* reads, const float* rs_W, const float* rs_b,
    const float* bn_g, const float* bn_b, const float* bn_m, const float* bn_v,
    const float* st_W, const float* st_b, const float* fo_W, const float* fo_b,
    float* out) {
  __shared__ float xm[64];
  __shared__ float h1[512];
  __shared__ float h2[128];
  int b = blockIdx.x, tid = threadIdx.x;
  if (tid < 64) {
    float s = 0.0f;
    for (int mem = 0; mem < 6; ++mem) s += reads[((long long)mem * 512 + b) * 64 + tid];
    xm[tid] = s * (1.0f / 6.0f);
  }
  __syncthreads();
  {
    float a = rs_b[tid];
    for (int d = 0; d < 64; ++d) a += xm[d] * rs_W[d * 512 + tid];
    a = fmaxf(a, 0.0f);
    a = bn_g[tid] * (a - bn_m[tid]) * rsqrtf(bn_v[tid] + 1e-3f) + bn_b[tid];
    h1[tid] = a;
  }
  __syncthreads();
  if (tid < 128) {
    float a = st_b[tid];
    for (int k = 0; k < 512; ++k) a += h1[k] * st_W[k * 128 + tid];
    h2[tid] = a;
  }
  __syncthreads();
  if (tid == 0) {
    float a = fo_b[0];
    for (int k = 0; k < 128; ++k) a += h2[k] * fo_W[k];
    out[b] = sigm(a);
  }
}

// ---------------------------------------------------------------- host side
static inline void launch_gemm(const GemmP& p, int z, hipStream_t s) {
  dim3 g((unsigned)((p.N + 127) / 128), (unsigned)((p.M + 127) / 128), (unsigned)z);
  if ((p.M % 128 == 0) && (p.N % 128 == 0))
    gemm_f16_wmma<true><<<g, dim3(256), 0, s>>>(p);
  else
    gemm_f16_wmma<false><<<g, dim3(256), 0, s>>>(p);
}

extern "C" void kernel_launch(void* const* d_in, const int* in_sizes, int n_in,
                              void* d_out, int out_size, void* d_ws, size_t ws_size,
                              hipStream_t stream) {
  (void)in_sizes; (void)n_in; (void)out_size; (void)ws_size;
  const int*   tokens = (const int*)  d_in[0];
  const float* emb    = (const float*)d_in[1];
  const float* l1f_Wi = (const float*)d_in[2];
  const float* l1f_Wh = (const float*)d_in[3];
  const float* l1f_b  = (const float*)d_in[4];
  const float* l1b_Wi = (const float*)d_in[5];
  const float* l1b_Wh = (const float*)d_in[6];
  const float* l1b_b  = (const float*)d_in[7];
  const float* l2f_Wi = (const float*)d_in[8];
  const float* l2f_Wh = (const float*)d_in[9];
  const float* l2f_b  = (const float*)d_in[10];
  const float* l2b_Wi = (const float*)d_in[11];
  const float* l2b_Wh = (const float*)d_in[12];
  const float* l2b_b  = (const float*)d_in[13];
  // d_in[14..17] = Wq,bq,Wk,bk : dead code (softmax over length-1 axis == 1)
  const float* Wv   = (const float*)d_in[18];
  const float* bv   = (const float*)d_in[19];
  const float* Wo   = (const float*)d_in[20];
  const float* bo   = (const float*)d_in[21];
  const float* mq_W = (const float*)d_in[22];
  const float* mq_b = (const float*)d_in[23];
  const float* rs_W = (const float*)d_in[36];
  const float* rs_b = (const float*)d_in[37];
  const float* bn_g = (const float*)d_in[38];
  const float* bn_b = (const float*)d_in[39];
  const float* bn_m = (const float*)d_in[40];
  const float* bn_v = (const float*)d_in[41];
  const float* st_W = (const float*)d_in[42];
  const float* st_b = (const float*)d_in[43];
  const float* fo_W = (const float*)d_in[44];
  const float* fo_b = (const float*)d_in[45];
  float* out = (float*)d_out;

  // ---- workspace carve (256B aligned)
  char* wsp = (char*)d_ws;
  auto alloc_h = [&](long long n) { _Float16* p = (_Float16*)wsp; wsp += ((n * 2 + 255) / 256) * 256; return p; };
  auto alloc_f = [&](long long n) { float* p = (float*)wsp; wsp += ((n * 4 + 255) / 256) * 256; return p; };

  _Float16* WT_L1WI = alloc_h(2LL * 2048 * 64);
  _Float16* WT_L1WH = alloc_h(2LL * 2048 * 512);
  _Float16* WT_L2WI = alloc_h(2LL * 2048 * 1024);
  _Float16* WT_L2WH = alloc_h(2LL * 2048 * 512);
  _Float16* WT_WV   = alloc_h(512LL * 1024);
  _Float16* WT_WO   = alloc_h(1024LL * 512);
  _Float16* WT_MQ   = alloc_h(64LL * 1024);
  _Float16* XH      = alloc_h(32768LL * 64);        // embedded tokens, f16
  _Float16* XW      = alloc_h(2LL * 32768 * 2048);  // input projections per dir
  _Float16* HSH     = alloc_h(32768LL * 1024);      // layer-1 seq out = layer-2 in
  _Float16* HBUF    = alloc_h(2LL * 512 * 512);     // h state per dir (f16)
  _Float16* HCAT    = alloc_h(512LL * 1024);        // final [hf, hb]
  _Float16* CTXH    = alloc_h(512LL * 512);
  _Float16* ATTNH   = alloc_h(512LL * 1024);
  float* ZBUF  = alloc_f(2LL * 512 * 2048);         // step gate pre-activations
  float* CBUF  = alloc_f(2LL * 512 * 512);          // c state per dir (f32)
  float* QUERY = alloc_f(512LL * 64);
  float* READS = alloc_f(6LL * 512 * 64);

  // ---- weight transpose + f16 convert
  auto tr = [&](const float* in, _Float16* o, int R, int C) {
    dim3 g((unsigned)((C + 31) / 32), (unsigned)((R + 31) / 32));
    transpose_conv<<<g, dim3(32, 8), 0, stream>>>(in, o, R, C);
  };
  tr(l1f_Wi, WT_L1WI,                 64, 2048);
  tr(l1b_Wi, WT_L1WI + 2048LL * 64,   64, 2048);
  tr(l1f_Wh, WT_L1WH,                512, 2048);
  tr(l1b_Wh, WT_L1WH + 2048LL * 512, 512, 2048);
  tr(l2f_Wi, WT_L2WI,                1024, 2048);
  tr(l2b_Wi, WT_L2WI + 2048LL * 1024, 1024, 2048);
  tr(l2f_Wh, WT_L2WH,                512, 2048);
  tr(l2b_Wh, WT_L2WH + 2048LL * 512, 512, 2048);
  tr(Wv,   WT_WV, 1024, 512);
  tr(Wo,   WT_WO,  512, 1024);
  tr(mq_W, WT_MQ, 1024, 64);

  // ---- embedding gather -> f16
  embed_kernel<<<dim3(8192), dim3(256), 0, stream>>>(tokens, emb, XH, 32768);

  const float* biasesWi[2][2] = { { l1f_b, l1b_b }, { l2f_b, l2b_b } };

  for (int layer = 0; layer < 2; ++layer) {
    // zero h/c states
    zero_u32<<<dim3((262144 + 255) / 256), dim3(256), 0, stream>>>((uint32_t*)HBUF, 262144);
    zero_u32<<<dim3((524288 + 255) / 256), dim3(256), 0, stream>>>((uint32_t*)CBUF, 524288);

    // input projections: XW[dir] = X @ Wi + b   (M=32768 fully parallel)
    for (int d = 0; d < 2; ++d) {
      GemmP p{};
      if (layer == 0) { p.A = XH;  p.ldA = 64;   p.K = 64;
                        p.Bt = WT_L1WI + (long long)d * 2048 * 64;   p.ldB = 64; }
      else            { p.A = HSH; p.ldA = 1024; p.K = 1024;
                        p.Bt = WT_L2WI + (long long)d * 2048 * 1024; p.ldB = 1024; }
      p.bias = biasesWi[layer][d];
      p.Ch = XW + (long long)d * 32768 * 2048; p.ldC = 2048;
      p.M = 32768; p.N = 2048;
      launch_gemm(p, 1, stream);
    }

    // recurrence: both directions batched over blockIdx.z
    _Float16* WhT = (layer == 0) ? WT_L1WH : WT_L2WH;
    for (int s = 0; s < 64; ++s) {
      GemmP p{};
      p.A = HBUF;  p.ldA = 512; p.sA = 512LL * 512;
      p.Bt = WhT;  p.ldB = 512; p.sB = 2048LL * 512;
      p.add = XW + (long long)s * 2048; p.ldAdd = 64LL * 2048;
      p.sAdd = (32768LL + 63 - 2 * s) * 2048;   // dir1 base at t = 63 - s
      p.Cf = ZBUF; p.sCf = 512LL * 2048; p.ldC = 2048;
      p.M = 512; p.N = 2048; p.K = 512;
      launch_gemm(p, 2, stream);

      if (layer == 0)
        lstm_gate<<<dim3(1024, 2), dim3(256), 0, stream>>>(ZBUF, CBUF, HBUF, HSH, s, 63 - s, 1);
      else
        lstm_gate<<<dim3(1024, 2), dim3(256), 0, stream>>>(ZBUF, CBUF, HBUF, HCAT, s, 63 - s, 2);
    }
  }

  // ---- head: ctx = hcat@Wv+bv ; attn = ctx@Wo+bo ; query = attn@mq_W+mq_b
  {
    GemmP p{};
    p.A = HCAT; p.ldA = 1024; p.Bt = WT_WV; p.ldB = 1024; p.bias = bv;
    p.Ch = CTXH; p.ldC = 512; p.M = 512; p.N = 512; p.K = 1024;
    launch_gemm(p, 1, stream);
  }
  {
    GemmP p{};
    p.A = CTXH; p.ldA = 512; p.Bt = WT_WO; p.ldB = 512; p.bias = bo;
    p.Ch = ATTNH; p.ldC = 1024; p.M = 512; p.N = 1024; p.K = 512;
    launch_gemm(p, 1, stream);
  }
  {
    GemmP p{};
    p.A = ATTNH; p.ldA = 1024; p.Bt = WT_MQ; p.ldB = 1024; p.bias = mq_b;
    p.Cf = QUERY; p.ldC = 64; p.M = 512; p.N = 64; p.K = 1024;
    launch_gemm(p, 1, stream);
  }

  // ---- six content-based memory reads
  MemsP mp;
  static const int memM[6] = { 100, 1000, 500, 100, 10000, 500 };
  for (int i = 0; i < 6; ++i) {
    mp.K[i] = (const float*)d_in[24 + 2 * i];
    mp.V[i] = (const float*)d_in[25 + 2 * i];
    mp.M[i] = memM[i];
  }
  mem_read_kernel<<<dim3(512, 6), dim3(64), 0, stream>>>(QUERY, mp, READS);

  // ---- fused tail
  tail_kernel<<<dim3(512), dim3(512), 0, stream>>>(
      READS, rs_W, rs_b, bn_g, bn_b, bn_m, bn_v, st_W, st_b, fo_W, fo_b, out);
}
